// MultiDecoders_70489003262731
// MI455X (gfx1250) — matrix operand order repaired and checked by reading, not compiled, not run on hardware
//
#include <hip/hip_runtime.h>
#include <math.h>

// ---------------- CDNA5 WMMA types ----------------
typedef __attribute__((ext_vector_type(16))) __bf16        v16bf;
typedef __attribute__((ext_vector_type(8)))  float         v8f;
typedef __attribute__((ext_vector_type(4)))  unsigned int  u4v;
typedef __attribute__((ext_vector_type(8)))  unsigned int  u8v;

#define SEQ    4
#define TSTEPS 23
#define BATCH  16
#define HID    512
#define EMBD   512
#define VOCAB  32000
#define KX     1024   /* EMB + HID          */
#define G3     1536   /* 3*HID              */
#define KF     1536   /* EMB + 2*HID        */
#define OUTL   24
#define MROWS  (SEQ*TSTEPS)          /* 92 M-tiles (16 rows each)  */
#define NTILE_V (VOCAB/16)           /* 2000 N-tiles               */
#define MP     (MROWS/2)             /* 46 real M-tile pairs       */
#define MPP    48                    /* padded to a multiple of 8  */
#define NQ     (NTILE_V/4)           /* 500 N-tile quads           */

// ---- A operand: 16x32 bf16 tile, row = lane&15, chunks per ISA layout ----
static __device__ __forceinline__ v16bf load_a16(const __bf16* row, int k0, int khalf) {
    const u4v* p0 = (const u4v*)(row + k0 + khalf * 8);        // K = k0+{0..7}  (or +8..15)
    const u4v* p1 = (const u4v*)(row + k0 + 16 + khalf * 8);   // K = k0+{16..23} (or +24..31)
    u4v lo = *p0, hi = *p1;
    u8v t;
    t[0]=lo[0]; t[1]=lo[1]; t[2]=lo[2]; t[3]=lo[3];
    t[4]=hi[0]; t[5]=hi[1]; t[6]=hi[2]; t[7]=hi[3];
    return __builtin_bit_cast(v16bf, t);
}
// ---- B operand: 32x16 bf16 tile; weights stored N-major; same K chunks as A:
//      k0+8*khalf+{0..7} and k0+16+8*khalf+{0..7} ----
static __device__ __forceinline__ v16bf load_b16(const __bf16* colrow, int k0, int khalf) {
    const u4v* p0 = (const u4v*)(colrow + k0 + khalf * 8);
    const u4v* p1 = (const u4v*)(colrow + k0 + 16 + khalf * 8);
    u4v lo = *p0, hi = *p1;
    u8v t;
    t[0]=lo[0]; t[1]=lo[1]; t[2]=lo[2]; t[3]=lo[3];
    t[4]=hi[0]; t[5]=hi[1]; t[6]=hi[2]; t[7]=hi[3];
    return __builtin_bit_cast(v16bf, t);
}
static __device__ __forceinline__ v8f wmma_bf16(v16bf a, v16bf b, v8f c) {
    return __builtin_amdgcn_wmma_f32_16x16x32_bf16(false, a, false, b, (short)0, c, false, false);
}
static __device__ __forceinline__ float sigmoidf_(float x) {
    return 1.0f / (1.0f + __expf(-x));
}

// ================= Stage 0: f32 -> bf16 conversion =================
__global__ void cvt_bf16_kernel(const float* __restrict__ in, __bf16* __restrict__ out, size_t n) {
    size_t i = (size_t)blockIdx.x * blockDim.x + threadIdx.x;
    size_t stride = (size_t)gridDim.x * blockDim.x;
    for (; i < n; i += stride) out[i] = (__bf16)in[i];
}

// ================= Stage 1: gather emb / ctx / agg =================
// grid = S*T*B blocks; idx = (s*23+t)*16 + b
__global__ void gather_kernel(const int* __restrict__ input, const float* __restrict__ hidden,
                              const float* __restrict__ agg, const float* __restrict__ emb,
                              __bf16* __restrict__ xbuf, __bf16* __restrict__ feats) {
    int idx = blockIdx.x;
    int b  = idx & 15;
    int st = idx >> 4;
    int t  = st % TSTEPS;
    int s  = st / TSTEPS;
    int tok = input[(b * SEQ + s) * OUTL + t];
    const float* erow = emb + (size_t)tok * EMBD;
    const float* ctx  = (s == 0) ? (agg + (size_t)b * HID)
                                 : (hidden + ((size_t)s * BATCH + b) * HID);
    const float* ag   = agg + ((size_t)s * BATCH + b) * HID;
    size_t xo = (size_t)idx * KX;
    size_t fo = (size_t)idx * KF;
    for (int j = threadIdx.x; j < HID; j += blockDim.x) {
        float e = erow[j];
        xbuf [xo + j]         = (__bf16)e;
        feats[fo + j]         = (__bf16)e;        // feats[0:512]   = emb
        xbuf [xo + HID + j]   = (__bf16)ctx[j];   // x[512:1024]    = ctx
        feats[fo + 2*HID + j] = (__bf16)ag[j];    // feats[1024:..] = agg
    }
}

// ================= Stage 2: gi = x @ W_ih^T + b_ih  (t-parallel) =================
// grid = S*T * 96 wave-tiles, block = 1 wave
__global__ __launch_bounds__(32) void gi_gemm_kernel(const __bf16* __restrict__ xbuf,
                                                     const __bf16* __restrict__ Wihb,
                                                     const float* __restrict__ b_ih,
                                                     float* __restrict__ gi) {
    int blk = blockIdx.x;
    int nt  = blk % (G3 / 16);
    int st  = blk / (G3 / 16);
    int s   = st / TSTEPS;
    int lane  = threadIdx.x & 31;
    int col   = lane & 15;
    int khalf = lane >> 4;
    const __bf16* arow = xbuf + ((size_t)st * BATCH + col) * KX;
    const __bf16* brow = Wihb + (size_t)s * G3 * KX + (size_t)(nt * 16 + col) * KX;
    v8f c = {};
    for (int k0 = 0; k0 < KX; k0 += 32) {
        v16bf a = load_a16(arow, k0, khalf);
        v16bf b = load_b16(brow, k0, khalf);
        c = wmma_bf16(a, b, c);
    }
    int   g    = nt * 16 + col;
    float bias = b_ih[s * G3 + g];
#pragma unroll
    for (int r = 0; r < 8; r++) {
        int row = r + khalf * 8;  // batch index
        gi[((size_t)st * BATCH + row) * G3 + g] = c[r] + bias;
    }
}

// ================= Stage 3: sequential GRU scan (1 block / line) =================
// Each wave owns 4 j-tiles; computes gh_r/gh_z/gh_n WMMA tiles sharing the A
// operand (h in LDS as bf16), does gate math in registers, double-barriers
// around the h update.
__global__ __launch_bounds__(256) void gru_scan_kernel(const float* __restrict__ hidden,
                                                       const float* __restrict__ b_hh,
                                                       const float* __restrict__ gi,
                                                       const __bf16* __restrict__ Whhb,
                                                       __bf16* __restrict__ feats) {
    __shared__ float  hS[BATCH * HID];   // 32 KB
    __shared__ __bf16 hB[BATCH * HID];   // 16 KB
    int s    = blockIdx.x;
    int tid  = threadIdx.x;
    int wave = tid >> 5;
    int lane = tid & 31;
    int col  = lane & 15;
    int khalf = lane >> 4;

    for (int i = tid; i < BATCH * HID; i += 256) {
        float v = hidden[(size_t)s * BATCH * HID + i];
        hS[i] = v;
        hB[i] = (__bf16)v;
    }
    __syncthreads();

    const __bf16* Whh_s = Whhb + (size_t)s * G3 * HID;
    const float*  bhh_s = b_hh + (size_t)s * G3;

    for (int t = 0; t < TSTEPS; t++) {
        const float* giP = gi + (size_t)(s * TSTEPS + t) * BATCH * G3;
        float hn_keep[4][8];
#pragma unroll
        for (int ji = 0; ji < 4; ji++) {
            int jt = wave * 4 + ji;        // 0..31 j-tiles of 16 hidden units
            int j  = jt * 16 + col;
            const __bf16* arow = &hB[col * HID];
            const __bf16* br = Whh_s + (size_t)(j)          * HID;
            const __bf16* bz = Whh_s + (size_t)(HID   + j)  * HID;
            const __bf16* bn = Whh_s + (size_t)(2*HID + j)  * HID;
            v8f cr = {}, cz = {}, cn = {};
            for (int k0 = 0; k0 < HID; k0 += 32) {
                v16bf a = load_a16(arow, k0, khalf);
                cr = wmma_bf16(a, load_b16(br, k0, khalf), cr);
                cz = wmma_bf16(a, load_b16(bz, k0, khalf), cz);
                cn = wmma_bf16(a, load_b16(bn, k0, khalf), cn);
            }
            float bhr = bhh_s[j], bhz = bhh_s[HID + j], bhn = bhh_s[2*HID + j];
#pragma unroll
            for (int r = 0; r < 8; r++) {
                int b = r + khalf * 8;
                const float* gib = giP + (size_t)b * G3;
                float rr = sigmoidf_(gib[j]        + cr[r] + bhr);
                float zz = sigmoidf_(gib[HID + j]  + cz[r] + bhz);
                float nn = tanhf   (gib[2*HID + j] + rr * (cn[r] + bhn));
                hn_keep[ji][r] = (1.0f - zz) * nn + zz * hS[b * HID + j];
            }
        }
        __syncthreads();   // all reads of hB/hS done before updates
        __bf16* fB = feats + (size_t)(s * TSTEPS + t) * BATCH * KF;
#pragma unroll
        for (int ji = 0; ji < 4; ji++) {
            int j = (wave * 4 + ji) * 16 + col;
#pragma unroll
            for (int r = 0; r < 8; r++) {
                int b = r + khalf * 8;
                float hn = hn_keep[ji][r];
                hS[b * HID + j] = hn;
                hB[b * HID + j] = (__bf16)hn;
                fB[(size_t)b * KF + HID + j] = (__bf16)hn;   // feats[512:1024] = h_t
            }
        }
        __syncthreads();
    }
}

// ================= Stage 4: logits = feats @ fc_W^T + fc_b =================
// 2M x 4N register-blocked wave tiles (8 WMMAs / 6KB loads, ~21 FLOP/B).
// Task decode is B-sharing: the 8 waves of a block use the SAME N-quad
// (fc_W panels hit in WGP$) and 8 different M-pairs (feats is L2-resident).
// M-pair space padded 46->48 with a wave-uniform early-out so sharing is exact.
__global__ __launch_bounds__(256) void fc_gemm_kernel(const __bf16* __restrict__ feats,
                                                      const __bf16* __restrict__ fcWb,
                                                      const float* __restrict__ fc_b,
                                                      float* __restrict__ out) {
    int  wave  = threadIdx.x >> 5;
    int  lane  = threadIdx.x & 31;
    int  col   = lane & 15;
    int  khalf = lane >> 4;
    long task = (long)blockIdx.x * 8 + wave;     // 0 .. 48*500-1
    int  mp   = (int)(task % MPP);               // M-tile pair (0..47)
    int  nq   = (int)(task / MPP);               // N-tile quad, uniform per block
    if (mp >= MP) return;                        // wave-uniform: EXEC all-1s inside

    const __bf16* arow0 = feats + ((size_t)(mp * 2 + 0) * BATCH + col) * KF;
    const __bf16* arow1 = feats + ((size_t)(mp * 2 + 1) * BATCH + col) * KF;
    const __bf16* brow0 = fcWb + (size_t)((nq * 4 + 0) * 16 + col) * KF;
    const __bf16* brow1 = fcWb + (size_t)((nq * 4 + 1) * 16 + col) * KF;
    const __bf16* brow2 = fcWb + (size_t)((nq * 4 + 2) * 16 + col) * KF;
    const __bf16* brow3 = fcWb + (size_t)((nq * 4 + 3) * 16 + col) * KF;

    v8f acc[2][4];
#pragma unroll
    for (int m = 0; m < 2; m++)
#pragma unroll
        for (int j = 0; j < 4; j++) acc[m][j] = (v8f){};

    for (int k0 = 0; k0 < KF; k0 += 32) {
        // prefetch the streamed fc_W panels ahead (lowers to global_prefetch_b8)
        __builtin_prefetch(brow0 + k0 + 256, 0, 0);
        __builtin_prefetch(brow1 + k0 + 256, 0, 0);
        __builtin_prefetch(brow2 + k0 + 256, 0, 0);
        __builtin_prefetch(brow3 + k0 + 256, 0, 0);
        v16bf a0 = load_a16(arow0, k0, khalf);
        v16bf a1 = load_a16(arow1, k0, khalf);
        v16bf b0 = load_b16(brow0, k0, khalf);
        acc[0][0] = wmma_bf16(a0, b0, acc[0][0]);
        acc[1][0] = wmma_bf16(a1, b0, acc[1][0]);
        v16bf b1 = load_b16(brow1, k0, khalf);
        acc[0][1] = wmma_bf16(a0, b1, acc[0][1]);
        acc[1][1] = wmma_bf16(a1, b1, acc[1][1]);
        v16bf b2 = load_b16(brow2, k0, khalf);
        acc[0][2] = wmma_bf16(a0, b2, acc[0][2]);
        acc[1][2] = wmma_bf16(a1, b2, acc[1][2]);
        v16bf b3 = load_b16(brow3, k0, khalf);
        acc[0][3] = wmma_bf16(a0, b3, acc[0][3]);
        acc[1][3] = wmma_bf16(a1, b3, acc[1][3]);
    }

#pragma unroll
    for (int m = 0; m < 2; m++) {
        int mt = mp * 2 + m;
        int s = mt / TSTEPS, t = mt % TSTEPS;
#pragma unroll
        for (int j = 0; j < 4; j++) {
            int   v    = (nq * 4 + j) * 16 + col;
            float bias = fc_b[v];
#pragma unroll
            for (int r = 0; r < 8; r++) {
                int b = r + khalf * 8;   // batch index
                out[(((size_t)b * SEQ + s) * OUTL + (t + 1)) * VOCAB + v] = acc[m][j][r] + bias;
            }
        }
    }
}

// ================= Stage 5: one-hot first output position =================
__global__ void onehot_kernel(float* __restrict__ out) {
    size_t total = (size_t)BATCH * SEQ * VOCAB;
    size_t i = (size_t)blockIdx.x * blockDim.x + threadIdx.x;
    size_t stride = (size_t)gridDim.x * blockDim.x;
    for (; i < total; i += stride) {
        int v = (int)(i % VOCAB);
        size_t bs = i / VOCAB;
        int s = (int)(bs % SEQ);
        int b = (int)(bs / SEQ);
        out[(((size_t)b * SEQ + s) * OUTL) * VOCAB + v] = (v == 1) ? 1.0f : 0.0f;
    }
}

// ================= launch =================
extern "C" void kernel_launch(void* const* d_in, const int* in_sizes, int n_in,
                              void* d_out, int out_size, void* d_ws, size_t ws_size,
                              hipStream_t stream) {
    const int*   input  = (const int*)  d_in[0];
    const float* hidden = (const float*)d_in[1];
    const float* oagg   = (const float*)d_in[2];
    const float* emb    = (const float*)d_in[3];
    const float* W_ih   = (const float*)d_in[4];
    const float* W_hh   = (const float*)d_in[5];
    const float* b_ih   = (const float*)d_in[6];
    const float* b_hh   = (const float*)d_in[7];
    const float* fc_W   = (const float*)d_in[8];
    const float* fc_b   = (const float*)d_in[9];
    float* out = (float*)d_out;

    // ---- workspace carving (bytes, all 256B aligned) ----
    char* ws = (char*)d_ws;
    size_t o = 0;
    __bf16* fcWb  = (__bf16*)(ws + o); o += (size_t)VOCAB * KF * 2;          // 98.3 MB
    __bf16* Wihb  = (__bf16*)(ws + o); o += (size_t)SEQ * G3 * KX * 2;       // 12.6 MB
    __bf16* Whhb  = (__bf16*)(ws + o); o += (size_t)SEQ * G3 * HID * 2;      //  6.3 MB
    __bf16* xbuf  = (__bf16*)(ws + o); o += (size_t)MROWS * BATCH * KX * 2;  //  3.0 MB
    __bf16* feats = (__bf16*)(ws + o); o += (size_t)MROWS * BATCH * KF * 2;  //  4.5 MB
    float*  gi    = (float*) (ws + o); o += (size_t)MROWS * BATCH * G3 * 4;  //  9.0 MB

    // Stage 0: weight conversions
    cvt_bf16_kernel<<<4096, 256, 0, stream>>>(fc_W, fcWb, (size_t)VOCAB * KF);
    cvt_bf16_kernel<<<2048, 256, 0, stream>>>(W_ih, Wihb, (size_t)SEQ * G3 * KX);
    cvt_bf16_kernel<<<1024, 256, 0, stream>>>(W_hh, Whhb, (size_t)SEQ * G3 * HID);

    // Stage 1: gather emb/ctx/agg into bf16 buffers
    gather_kernel<<<MROWS * BATCH, 128, 0, stream>>>(input, hidden, oagg, emb, xbuf, feats);

    // Stage 2: gi GEMM (one wave per 16x16 tile)
    gi_gemm_kernel<<<MROWS * (G3 / 16), 32, 0, stream>>>(xbuf, Wihb, b_ih, gi);

    // Stage 3: sequential GRU scan, one block per line
    gru_scan_kernel<<<SEQ, 256, 0, stream>>>(hidden, b_hh, gi, Whhb, feats);

    // Stage 4: giant FC, B-sharing task decode (48*500 wave tasks / 8 per block)
    fc_gemm_kernel<<<(MPP * NQ) / 8, 256, 0, stream>>>(feats, fcWb, fc_b, out);

    // Stage 5: one-hot l=0 slab
    onehot_kernel<<<2048, 256, 0, stream>>>(out);
}
